// TruncatedCRF_90718299226738
// MI455X (gfx1250) — matrix operand order repaired and checked by reading, not compiled, not run on hardware
//
#include <hip/hip_runtime.h>
#include <hip/hip_bf16.h>

// Problem constants (from reference)
#define BB   64
#define TT   512
#define TM1  511          // T-1
#define LL   2048
#define DD   512
#define NROW (BB * TM1)   // 32704 = 511 * 64

typedef __attribute__((ext_vector_type(16))) __bf16 v16bf;
typedef __attribute__((ext_vector_type(8)))  float  v8f;

union ABu  { v16bf v; uint4 q[2]; };
union V8BF { uint4 q; __bf16 h[8]; };

#define NEG_INF (-1e30f)

// ---------------------------------------------------------------------------
// Kernel 1: one-shot f32 -> bf16 conversion of both embedding tables into d_ws
// ---------------------------------------------------------------------------
__global__ void __launch_bounds__(256)
cvt_bf16_kernel(const float* __restrict__ se, const float* __restrict__ te,
                __bf16* __restrict__ sb, __bf16* __restrict__ tb, int n) {
    int i = blockIdx.x * 256 + threadIdx.x;
    if (i < n) {
        sb[i] = (__bf16)se[i];
        tb[i] = (__bf16)te[i];
    }
}

// ---------------------------------------------------------------------------
// Kernel 2: fused gather-GEMM (bf16 WMMA, f32 acc) + per-lane online
// logsumexp.  One workgroup = 64 rows x all 2048 logit columns.
// Each wave owns 16 columns per chunk and computes FOUR 16x16 M-tiles that
// share the same B registers -> 4x B reuse (~1 GB total L2 traffic).
// No shuffles / no target capture in the hot loop; (m,s) merged at the end,
// target logit recomputed by a tiny f32 dot-product epilogue.
// ---------------------------------------------------------------------------
__global__ void __launch_bounds__(256)
crf_score_kernel(const int* __restrict__ labels,
                 const __bf16* __restrict__ srcb,   // [LL][DD] bf16
                 const __bf16* __restrict__ tgtb,   // [LL][DD] bf16
                 float* __restrict__ out) {
    extern __shared__ char smem_raw[];
    __bf16* abuf     = (__bf16*)smem_raw;                 // 64*512 bf16 = 65536 B
    int*    tcol_lds = (int*)  (smem_raw + 65536);        // 64 ints = 256 B
    float*  marr     = (float*)(smem_raw + 65792);        // 64*8 f32 = 2048 B
    float*  sarr     = (float*)(smem_raw + 67840);        // 2048 B
    // total 69888 B

    const int row0 = blockIdx.x * 64;
    const int tid  = threadIdx.x;
    const int wave = tid >> 5;
    const int lane = tid & 31;
    const int laneLo = lane & 15;
    const int hi     = lane >> 4;

    // ---- Stage A: gather 64 source-embedding rows (bf16) into LDS ----------
    for (int i = tid; i < 4096; i += 256) {
        int rrow = i >> 6;            // 64 vec8 per row
        int kpos = (i & 63) << 3;     // element offset within row
        int r = row0 + rrow;
        int b = r / TM1;
        int t = r - b * TM1;
        int lbl = labels[b * TT + t];
        const uint4* gp = (const uint4*)(srcb + (size_t)lbl * DD + kpos);
        *(uint4*)(abuf + rrow * DD + kpos) = *gp;
    }
    if (tid < 64) {
        int r = row0 + tid;
        int b = r / TM1;
        int t = r - b * TM1;
        tcol_lds[tid] = labels[b * TT + t + 1];   // target label per row
    }
    __syncthreads();

    // Per-lane online-logsumexp state for 32 owned rows (4 tiles x 8 VGPRs).
    float mrun[32], srun[32];
#pragma unroll
    for (int j = 0; j < 32; ++j) { mrun[j] = NEG_INF; srun[j] = 0.f; }

    // A operand base for this lane: row m = laneLo, K sub-offset k0 = hi*8.
    const __bf16* arow = abuf + laneLo * DD + hi * 8;

    // ---- Chunk loop: 8 waves x 16 cols = 128 cols/iter, 16 iters over L ----
#pragma unroll 1
    for (int chunk = 0; chunk < 16; ++chunk) {
        const int n = chunk * 128 + wave * 16 + laneLo;     // logit column
        // B operand: column n of (D x L) B == row n of target_embeddings.
        const __bf16* bcol = tgtb + (size_t)n * DD + hi * 8;

        v8f c0 = {}, c1 = {}, c2 = {}, c3 = {};
        // Partial unroll: keeps A loads kk-dependent so they CANNOT be
        // hoisted out of the chunk loop (avoids 512-VGPR hoist + spills).
#pragma unroll 2
        for (int kk = 0; kk < DD; kk += 32) {
            ABu bm;
            const uint4* bp = (const uint4*)(bcol + kk);
            bm.q[0] = bp[0];          // K = kk+kb .. +7
            bm.q[1] = bp[2];          // K = kk+kb+16 .. +23

            ABu a0, a1, a2, a3;
            const uint4* ap0 = (const uint4*)(arow + kk);
            const uint4* ap1 = (const uint4*)(arow + 16 * DD + kk);
            const uint4* ap2 = (const uint4*)(arow + 32 * DD + kk);
            const uint4* ap3 = (const uint4*)(arow + 48 * DD + kk);
            a0.q[0] = ap0[0]; a0.q[1] = ap0[2];   // K = kk+k0..+7, +16..+23
            a1.q[0] = ap1[0]; a1.q[1] = ap1[2];
            a2.q[0] = ap2[0]; a2.q[1] = ap2[2];
            a3.q[0] = ap3[0]; a3.q[1] = ap3[2];

            c0 = __builtin_amdgcn_wmma_f32_16x16x32_bf16(
                     false, a0.v, false, bm.v, (short)0, c0, false, false);
            c1 = __builtin_amdgcn_wmma_f32_16x16x32_bf16(
                     false, a1.v, false, bm.v, (short)0, c1, false, false);
            c2 = __builtin_amdgcn_wmma_f32_16x16x32_bf16(
                     false, a2.v, false, bm.v, (short)0, c2, false, false);
            c3 = __builtin_amdgcn_wmma_f32_16x16x32_bf16(
                     false, a3.v, false, bm.v, (short)0, c3, false, false);
        }

        // Per-lane online logsumexp fold (no cross-lane ops, single exp)
#pragma unroll
        for (int g = 0; g < 4; ++g) {
#pragma unroll
            for (int v = 0; v < 8; ++v) {
                float x  = (g == 0) ? c0[v] : (g == 1) ? c1[v]
                         : (g == 2) ? c2[v] : c3[v];
                int   j  = g * 8 + v;
                float mo = mrun[j];
                float d  = x - mo;
                float e  = __expf(-fabsf(d));      // exp(-|x-m|)
                bool  nm = d > 0.f;
                srun[j]  = nm ? (srun[j] * e + 1.f) : (srun[j] + e);
                mrun[j]  = nm ? x : mo;
            }
        }
    }

    // ---- Merge (m,s) across the 16 lanes of each half (uniform flow) -------
#pragma unroll
    for (int j = 0; j < 32; ++j) {
        float m = mrun[j], s = srun[j];
#pragma unroll
        for (int off = 1; off < 16; off <<= 1) {
            float mo = __shfl_xor(m, off, 32);
            float so = __shfl_xor(s, off, 32);
            float mn = fmaxf(m, mo);
            s = s * __expf(m - mn) + so * __expf(mo - mn);
            m = mn;
        }
        mrun[j] = m; srun[j] = s;
    }

    // One lane per half publishes this wave's per-row stats
    if (laneLo == 0) {
#pragma unroll
        for (int g = 0; g < 4; ++g)
#pragma unroll
            for (int v = 0; v < 8; ++v) {
                int r = g * 16 + v + 8 * hi;
                int j = g * 8 + v;
                marr[r * 8 + wave] = mrun[j];
                sarr[r * 8 + wave] = srun[j];
            }
    }
    __syncthreads();

    // ---- Epilogue: 64 threads, one row each --------------------------------
    if (tid < 64) {
        int r = tid;
        float m = NEG_INF;
#pragma unroll
        for (int w = 0; w < 8; ++w) m = fmaxf(m, marr[r * 8 + w]);
        float s = 0.f;
#pragma unroll
        for (int w = 0; w < 8; ++w)
            s += sarr[r * 8 + w] * __expf(marr[r * 8 + w] - m);

        // Target logit: f32 dot of gathered source row (LDS) with the
        // target-label embedding row (global bf16), vectorized 8-wide.
        int tc = tcol_lds[r];
        const __bf16* av = abuf + r * DD;
        const __bf16* bv = tgtb + (size_t)tc * DD;
        float acc = 0.f;
#pragma unroll 4
        for (int k = 0; k < DD; k += 8) {
            V8BF a, b;
            a.q = *(const uint4*)(av + k);
            b.q = *(const uint4*)(bv + k);
#pragma unroll
            for (int u = 0; u < 8; ++u)
                acc += (float)a.h[u] * (float)b.h[u];
        }
        out[row0 + r] = acc - m - __logf(s);
    }
}

// ---------------------------------------------------------------------------
extern "C" void kernel_launch(void* const* d_in, const int* in_sizes, int n_in,
                              void* d_out, int out_size, void* d_ws, size_t ws_size,
                              hipStream_t stream) {
    const int*   labels = (const int*)  d_in[0];   // (64, 512) int32
    const float* src_f  = (const float*)d_in[1];   // (2048, 512) f32
    const float* tgt_f  = (const float*)d_in[2];   // (2048, 512) f32
    float* out = (float*)d_out;                    // (64, 511) f32

    const int nElem = LL * DD;                     // 1,048,576 per table
    __bf16* src_b = (__bf16*)d_ws;                 // 2 MB
    __bf16* tgt_b = (__bf16*)((char*)d_ws + (size_t)nElem * sizeof(__bf16));

    cvt_bf16_kernel<<<(nElem + 255) / 256, 256, 0, stream>>>(
        src_f, tgt_f, src_b, tgt_b, nElem);

    const int smemBytes = 69888;                   // A-tile + stats arrays
    crf_score_kernel<<<NROW / 64, 256, smemBytes, stream>>>(
        labels, src_b, tgt_b, out);
}
